// CausalSelfAttention_43087111913728
// MI455X (gfx1250) — matrix operand order repaired and checked by reading, not compiled, not run on hardware
//
#include <hip/hip_runtime.h>

// ---------------- types ----------------
typedef __bf16 bf16;
typedef bf16  v16bf __attribute__((ext_vector_type(16)));
typedef bf16  v8bf  __attribute__((ext_vector_type(8)));
typedef float v8f   __attribute__((ext_vector_type(8)));

#define B_  2
#define S_  2048
#define D_  1024
#define H_  16
#define HD_ 64

// ---------------- WMMA fragment helpers (CDNA5 layouts, wave32) ----------------
// A-matrix 16x32 bf16: lane L (0-15) holds row M=L, K in {0-7,16-23};
// lanes 16-31 hold M=L-16, K in {8-15,24-31}. elem j -> k = kb + (j&7) + 16*(j>>3).
__device__ __forceinline__ v16bf load_a(const bf16* __restrict__ base, int ld, int kk) {
  const int lane = threadIdx.x & 31;
  const int m  = lane & 15;
  const int kb = (lane >> 4) * 8;
  const bf16* p = base + (size_t)m * ld + kk + kb;
  v8bf lo = *(const v8bf*)(p);        // 16B aligned
  v8bf hi = *(const v8bf*)(p + 16);   // 16B aligned
  v16bf a;
#pragma unroll
  for (int j = 0; j < 8; ++j) { a[j] = lo[j]; a[8 + j] = hi[j]; }
  return a;
}

// B-matrix 32x16 bf16 (K x N): lane L holds column N = n0+(L&15); elem j -> k = kk + 8*(L>>4) + (j&7) + 16*(j>>3).
__device__ __forceinline__ v16bf load_b(const bf16* __restrict__ base, int ldn,
                                        int kk, int n0) {
  const int lane = threadIdx.x & 31;
  const int kb = (lane >> 4) * 8;
  const bf16* p = base + (size_t)(kk + kb) * ldn + n0 + (lane & 15);
  v16bf b;
#pragma unroll
  for (int j = 0; j < 8; ++j) { b[j] = p[(size_t)j * ldn]; b[8 + j] = p[(size_t)(16 + j) * ldn]; }
  return b;
}

__device__ __forceinline__ v8f wmma_bf16(v16bf a, v16bf b, v8f c) {
  return __builtin_amdgcn_wmma_f32_16x16x32_bf16(false, a, false, b, (short)0, c,
                                                 false, false);
}

// ---------------- 32x64 GEMM core, single-buffered (spill-free) ----------------
// acc[ms][t] covers rows m0+16*ms..+15, cols n0+16*t..+15 of  A(32xD) @ WT(DxN).
__device__ __forceinline__ void gemm_32x64(const bf16* __restrict__ arow,
                                           const bf16* __restrict__ WT,
                                           int n0, v8f acc[2][4]) {
#pragma unroll 1
  for (int kk = 0; kk < D_; kk += 32) {
    // issue all fragment loads up front (one clause), WMMAs drain behind staged waits
    v16bf a0 = load_a(arow, D_, kk);
    v16bf a1 = load_a(arow + (size_t)16 * D_, D_, kk);
    v16bf b0 = load_b(WT, D_, kk, n0);
    v16bf b1 = load_b(WT, D_, kk, n0 + 16);
    v16bf b2 = load_b(WT, D_, kk, n0 + 32);
    v16bf b3 = load_b(WT, D_, kk, n0 + 48);
    acc[0][0] = wmma_bf16(a0, b0, acc[0][0]);
    acc[0][1] = wmma_bf16(a0, b1, acc[0][1]);
    acc[0][2] = wmma_bf16(a0, b2, acc[0][2]);
    acc[0][3] = wmma_bf16(a0, b3, acc[0][3]);
    acc[1][0] = wmma_bf16(a1, b0, acc[1][0]);
    acc[1][1] = wmma_bf16(a1, b1, acc[1][1]);
    acc[1][2] = wmma_bf16(a1, b2, acc[1][2]);
    acc[1][3] = wmma_bf16(a1, b3, acc[1][3]);
  }
}

// ---------------- kernel 0a: weight transpose + bf16 convert ----------------
__global__ void wt_kernel(const float* __restrict__ Wq, const float* __restrict__ Wk,
                          const float* __restrict__ Wv, const float* __restrict__ Wo,
                          bf16* __restrict__ WqT, bf16* __restrict__ WkT,
                          bf16* __restrict__ WvT, bf16* __restrict__ WoT) {
  size_t i = (size_t)blockIdx.x * blockDim.x + threadIdx.x;
  if (i >= (size_t)4 * D_ * D_) return;
  int w = (int)(i / ((size_t)D_ * D_));
  int r = (int)(i % ((size_t)D_ * D_));
  int k = r / D_;
  int n = r % D_;
  const float* W = (w == 0) ? Wq : (w == 1) ? Wk : (w == 2) ? Wv : Wo;
  bf16* WT       = (w == 0) ? WqT : (w == 1) ? WkT : (w == 2) ? WvT : WoT;
  WT[(size_t)k * D_ + n] = (bf16)W[(size_t)n * D_ + k];
}

// ---------------- kernel 0b: x -> bf16 (so GEMM A-fragments are pure b128 loads) ----
__global__ void xcvt_kernel(const float* __restrict__ x, bf16* __restrict__ xbf) {
  size_t i = (size_t)blockIdx.x * blockDim.x + threadIdx.x;
  if (i >= (size_t)B_ * S_ * D_) return;
  xbf[i] = (bf16)x[i];
}

// ---------------- kernel 1: fused QKV projection ----------------
// One wave = one 32x64 tile of one of {Q,K,V}. Q,V stored [b,h,s,hd]; K stored [b,h,hd,s].
__global__ void qkv_proj_kernel(const bf16* __restrict__ xbf,
                                const bf16* __restrict__ WqT, const bf16* __restrict__ WkT,
                                const bf16* __restrict__ WvT,
                                const float* __restrict__ bq, const float* __restrict__ bk,
                                const float* __restrict__ bv,
                                bf16* __restrict__ Qd, bf16* __restrict__ Kt,
                                bf16* __restrict__ Vd) {
  const int tiles_n = D_ / 64;                       // 16
  const int tiles_m = (B_ * S_) / 32;                // 128
  const int tiles_per_mat = tiles_m * tiles_n;       // 2048
  int wave = blockIdx.x * (blockDim.x >> 5) + (threadIdx.x >> 5);
  int sel = wave / tiles_per_mat;                    // 0=Q 1=K 2=V
  if (sel >= 3) return;
  int t  = wave % tiles_per_mat;
  int m0 = (t / tiles_n) * 32;
  int n0 = (t % tiles_n) * 64;

  const bf16*  WT   = (sel == 0) ? WqT : (sel == 1) ? WkT : WvT;
  const float* bias = (sel == 0) ? bq  : (sel == 1) ? bk  : bv;

  v8f acc[2][4] = {};
  gemm_32x64(xbf + (size_t)m0 * D_, WT, n0, acc);

  const int lane = threadIdx.x & 31;
  const int nl = lane & 15, hl = lane >> 4;
#pragma unroll
  for (int tt = 0; tt < 4; ++tt) {
    int n = n0 + tt * 16 + nl;
    float bb = bias[n];
    int h = n >> 6, hd = n & 63;
#pragma unroll
    for (int ms = 0; ms < 2; ++ms)
#pragma unroll
      for (int v = 0; v < 8; ++v) {
        int row = m0 + 16 * ms + v + 8 * hl;
        int b   = row >> 11;           // / S_
        int s   = row & (S_ - 1);
        bf16 val = (bf16)(acc[ms][tt][v] + bb);
        if (sel == 0)      Qd[(((size_t)(b * H_ + h)) * S_ + s) * HD_ + hd] = val;
        else if (sel == 2) Vd[(((size_t)(b * H_ + h)) * S_ + s) * HD_ + hd] = val;
        else               Kt[(((size_t)(b * H_ + h)) * HD_ + hd) * S_ + s] = val;
      }
  }
}

// ---------------- kernel 2: flash-style causal attention ----------------
// One wave = one (b,h, 32-query tile) as two 16-row M-subtiles sharing K/V fragments.
__global__ void attn_kernel(const bf16* __restrict__ Qd, const bf16* __restrict__ Kt,
                            const bf16* __restrict__ Vd, bf16* __restrict__ ctx) {
  __shared__ bf16 lds_p[8][32 * 32];                 // per-wave P staging (2KB each)
  const int wslot = threadIdx.x >> 5;
  int wave = blockIdx.x * (blockDim.x >> 5) + wslot;
  const int qtiles = S_ / 32;                        // 64
  int bh = wave / qtiles;
  if (bh >= B_ * H_) return;
  int q0 = (wave % qtiles) * 32;

  const int lane = threadIdx.x & 31;
  const int nl = lane & 15, hl = lane >> 4;

  const bf16* Qb  = Qd + (size_t)bh * S_ * HD_;
  const bf16* Ktb = Kt + (size_t)bh * HD_ * S_;
  const bf16* Vb  = Vd + (size_t)bh * S_ * HD_;

  // Q A-fragments: [m-subtile][hd-chunk], kept in registers (32 VGPRs)
  v16bf aq[2][2];
#pragma unroll
  for (int ms = 0; ms < 2; ++ms)
#pragma unroll
    for (int c = 0; c < 2; ++c)
      aq[ms][c] = load_a(Qb + (size_t)(q0 + 16 * ms) * HD_, HD_, c * 32);

  v8f acc[2][4] = {};
  float rowmax[2][8], rowsum[2][8];
#pragma unroll
  for (int ms = 0; ms < 2; ++ms)
#pragma unroll
    for (int v = 0; v < 8; ++v) { rowmax[ms][v] = -3.0e38f; rowsum[ms][v] = 0.0f; }

  bf16* myp = &lds_p[wslot][0];

#pragma unroll 1
  for (int k0 = 0; k0 <= q0 + 31; k0 += 32) {        // causal: skip fully-masked chunks
    // ---- scores: 32q x 32k; key N-tiles processed sequentially (2 K-frags live) ----
    v8f sc[2][2] = {};
#pragma unroll
    for (int nt = 0; nt < 2; ++nt) {
      v16bf bk0 = load_b(Ktb, S_, 0,  k0 + nt * 16);
      v16bf bk1 = load_b(Ktb, S_, 32, k0 + nt * 16);
#pragma unroll
      for (int ms = 0; ms < 2; ++ms) {
        sc[ms][nt] = wmma_bf16(aq[ms][0], bk0, sc[ms][nt]);
        sc[ms][nt] = wmma_bf16(aq[ms][1], bk1, sc[ms][nt]);
      }
    }

    // ---- scale, causal mask, online softmax (fp32), P -> LDS; per-ms frees regs ----
#pragma unroll
    for (int ms = 0; ms < 2; ++ms) {
#pragma unroll
      for (int v = 0; v < 8; ++v) {
        float s0 = sc[ms][0][v] * 0.125f;            // 1/sqrt(64)
        float s1 = sc[ms][1][v] * 0.125f;
        int qi = q0 + 16 * ms + v + 8 * hl;
        if (k0 + nl      > qi) s0 = -1.0e30f;
        if (k0 + 16 + nl > qi) s1 = -1.0e30f;
        float bm = fmaxf(s0, s1);
        bm = fmaxf(bm, __shfl_xor(bm, 1, 32));
        bm = fmaxf(bm, __shfl_xor(bm, 2, 32));
        bm = fmaxf(bm, __shfl_xor(bm, 4, 32));
        bm = fmaxf(bm, __shfl_xor(bm, 8, 32));
        float nm = fmaxf(rowmax[ms][v], bm);
        float al = __expf(rowmax[ms][v] - nm);
        rowmax[ms][v] = nm;
        float p0 = __expf(s0 - nm);
        float p1 = __expf(s1 - nm);
        float bs = p0 + p1;
        bs += __shfl_xor(bs, 1, 32);
        bs += __shfl_xor(bs, 2, 32);
        bs += __shfl_xor(bs, 4, 32);
        bs += __shfl_xor(bs, 8, 32);
        rowsum[ms][v] = rowsum[ms][v] * al + bs;
#pragma unroll
        for (int t = 0; t < 4; ++t) acc[ms][t][v] *= al;
        int m = 16 * ms + v + 8 * hl;                // P row
        myp[m * 32 + nl]      = (bf16)p0;
        myp[m * 32 + 16 + nl] = (bf16)p1;
      }
    }
    asm volatile("s_wait_dscnt 0" ::: "memory");     // LDS stores visible to reload

    // ---- reload P (2x 16x32) in A-fragment layout ----
    v16bf pa[2];
#pragma unroll
    for (int ms = 0; ms < 2; ++ms) {
      const int mA = lane & 15;
      const int kb = hl * 8;
      const bf16* pp = myp + (16 * ms + mA) * 32 + kb;
      v8bf lo = *(const v8bf*)pp;
      v8bf hi = *(const v8bf*)(pp + 16);
#pragma unroll
      for (int j = 0; j < 8; ++j) { pa[ms][j] = lo[j]; pa[ms][8 + j] = hi[j]; }
    }
    // ---- ctx += P(32x32) @ V(32x64); one live V fragment at a time ----
#pragma unroll
    for (int t = 0; t < 4; ++t) {
      v16bf bv = load_b(Vb + (size_t)k0 * HD_, HD_, 0, t * 16);
      acc[0][t] = wmma_bf16(pa[0], bv, acc[0][t]);
      acc[1][t] = wmma_bf16(pa[1], bv, acc[1][t]);
    }
  }

  // ---- epilogue: normalize, store ctx as [b, s, h*64+hd] bf16 ----
  int b = bh / H_, h = bh % H_;
#pragma unroll
  for (int ms = 0; ms < 2; ++ms)
#pragma unroll
    for (int t = 0; t < 4; ++t)
#pragma unroll
      for (int v = 0; v < 8; ++v) {
        int m = 16 * ms + v + 8 * hl;
        float val = acc[ms][t][v] / rowsum[ms][v];
        ctx[((size_t)b * S_ + q0 + m) * D_ + h * HD_ + t * 16 + nl] = (bf16)val;
      }
}

// ---------------- kernel 3: output projection (bf16 ctx -> fp32 out) ----------------
__global__ void out_proj_kernel(const bf16* __restrict__ ctx, const bf16* __restrict__ WoT,
                                const float* __restrict__ bo, float* __restrict__ out) {
  const int tiles_n = D_ / 64;                       // 16
  const int tiles_m = (B_ * S_) / 32;                // 128
  int wave = blockIdx.x * (blockDim.x >> 5) + (threadIdx.x >> 5);
  if (wave >= tiles_m * tiles_n) return;
  int m0 = (wave / tiles_n) * 32;
  int n0 = (wave % tiles_n) * 64;

  v8f acc[2][4] = {};
  gemm_32x64(ctx + (size_t)m0 * D_, WoT, n0, acc);

  const int lane = threadIdx.x & 31;
  const int nl = lane & 15, hl = lane >> 4;
#pragma unroll
  for (int t = 0; t < 4; ++t) {
    float bias = bo[n0 + t * 16 + nl];
#pragma unroll
    for (int ms = 0; ms < 2; ++ms)
#pragma unroll
      for (int v = 0; v < 8; ++v) {
        int row = m0 + 16 * ms + v + 8 * hl;
        out[(size_t)row * D_ + n0 + t * 16 + nl] = acc[ms][t][v] + bias;
      }
  }
}

// ---------------- launch ----------------
extern "C" void kernel_launch(void* const* d_in, const int* in_sizes, int n_in,
                              void* d_out, int out_size, void* d_ws, size_t ws_size,
                              hipStream_t stream) {
  (void)in_sizes; (void)n_in; (void)out_size; (void)ws_size;
  const float* x  = (const float*)d_in[0];
  // d_in[1] = mask (implicit via causal loop)
  const float* Wq = (const float*)d_in[2];
  const float* bq = (const float*)d_in[3];
  const float* Wk = (const float*)d_in[4];
  const float* bk = (const float*)d_in[5];
  const float* Wv = (const float*)d_in[6];
  const float* bv = (const float*)d_in[7];
  const float* Wo = (const float*)d_in[8];
  const float* bo = (const float*)d_in[9];
  float* out = (float*)d_out;

  bf16* ws = (bf16*)d_ws;
  const size_t DD = (size_t)D_ * D_;
  const size_t QL = (size_t)B_ * H_ * S_ * HD_;   // == B_*S_*D_
  bf16* WqT = ws;
  bf16* WkT = WqT + DD;
  bf16* WvT = WkT + DD;
  bf16* WoT = WvT + DD;
  bf16* xbf = WoT + DD;
  bf16* Qd  = xbf + QL;
  bf16* Kt  = Qd + QL;
  bf16* Vd  = Kt + QL;
  bf16* ctx = Vd + QL;

  {
    int n = 4 * D_ * D_;
    wt_kernel<<<(n + 255) / 256, 256, 0, stream>>>(Wq, Wk, Wv, Wo, WqT, WkT, WvT, WoT);
  }
  {
    int n = B_ * S_ * D_;
    xcvt_kernel<<<(n + 255) / 256, 256, 0, stream>>>(x, xbf);
  }
  {
    int waves = 3 * ((B_ * S_) / 32) * (D_ / 64);   // 6144
    qkv_proj_kernel<<<waves / 8, 256, 0, stream>>>(xbf, WqT, WkT, WvT, bq, bk, bv, Qd, Kt, Vd);
  }
  {
    int waves = B_ * H_ * (S_ / 32);                // 2048
    attn_kernel<<<waves / 8, 256, 0, stream>>>(Qd, Kt, Vd, ctx);
  }
  {
    int waves = ((B_ * S_) / 32) * (D_ / 64);       // 2048
    out_proj_kernel<<<waves / 8, 256, 0, stream>>>(ctx, WoT, bo, out);
  }
}
